// Interactor_66640712565228
// MI455X (gfx1250) — matrix operand order repaired and checked by reading, not compiled, not run on hardware
//
#include <hip/hip_runtime.h>
#include <hip/hip_bf16.h>

// Problem constants (match reference)
#define BB 64
#define TT 512
#define LL 48
#define HH 512   // DV == DS == H == 512

typedef __attribute__((ext_vector_type(16))) _Float16 v16h;
typedef __attribute__((ext_vector_type(8)))  _Float16 v8h;
typedef __attribute__((ext_vector_type(8)))  float    v8f;
typedef __attribute__((ext_vector_type(4)))  unsigned v4u;
typedef __attribute__((ext_vector_type(8)))  int      v8i;
typedef __attribute__((ext_vector_type(4)))  int      v4i;

static __device__ __forceinline__ int lane31() { return (int)(threadIdx.x & 31u); }

// ---- CDNA5 WMMA tile loads, contiguous k = 16*(lane/16) + s in A and B ----------------------
// A-matrix 16x32 f16, row-major source with row stride lda (halves).
// lanes 0-15: row m, K 0-15 ; lanes 16-31: row m, K 16-31.
template <typename TP>
static __device__ __forceinline__ v16h load_A16x32(const TP* A, int lda) {
  int l  = lane31();
  int m  = l & 15;
  int hs = (l >> 4) & 1;
  const TP* p = A + m * lda + hs * 16;
  union { v16h v; v8h h[2]; } u;
  u.h[0] = *(const v8h*)(p);
  u.h[1] = *(const v8h*)(p + 8);
  return u.v;
}

// B-matrix 32x16 f16 where B[k,n] = W[n0+n, k0+k], W row-major (stride ldw).
// lanes 0-15 hold K 0-15 of column n; lanes 16-31 hold K 16-31. One 32B run.
static __device__ __forceinline__ v16h load_B32x16(const _Float16* Wtile, int ldw) {
  int l    = lane31();
  int n    = l & 15;
  int koff = ((l >> 4) & 1) * 16;
  return *(const v16h*)(Wtile + n * ldw + koff);
}

static __device__ __forceinline__ v8f wmma16(v16h a, v16h b, v8f c) {
  // D = A x B + C ; f32 accumulate
  return __builtin_amdgcn_wmma_f32_16x16x32_f16(false, a, false, b, (short)0, c,
                                                false, false);
}

static __device__ __forceinline__ float sigmf(float x) {
  return 1.0f / (1.0f + expf(-x));
}

// ---- Tensor Data Mover: 2-D tile (rows x row_len f16) global -> LDS ----------
#if __has_builtin(__builtin_amdgcn_tensor_load_to_lds)
#define USE_TDM 1
static __device__ __forceinline__ void tdm_load_2d_f16(
    unsigned lds_byte_off, const _Float16* gsrc,
    unsigned rows, unsigned row_len, unsigned long long row_stride_elems)
{
  unsigned long long ga = (unsigned long long)(uintptr_t)gsrc;
  // D# group0 (ISA 8.3): count=1 | lds_addr | global_addr[56:0] | type=2
  v4u g0;
  g0[0] = 1u;
  g0[1] = lds_byte_off;
  g0[2] = (unsigned)(ga & 0xffffffffu);
  g0[3] = (unsigned)((ga >> 32) & 0x01ffffffu) | (2u << 30);
  // D# group1 (ISA 8.4): data_size=1(2B); tensor_dim0/1; tile_dim0/1; dim0 stride
  v8i g1;
  g1[0] = (int)(1u << 16);                                   // data_size = 2 bytes
  g1[1] = (int)((row_len & 0xffffu) << 16);                  // tensor_dim0[15:0]
  g1[2] = (int)((row_len >> 16) & 0xffffu)                   // tensor_dim0[31:16]
        | (int)((rows & 0xffffu) << 16);                     // tensor_dim1[15:0]
  g1[3] = (int)((row_len & 0xffffu) << 16);                  // tile_dim0 (dim1 hi = 0)
  g1[4] = (int)(rows & 0xffffu);                             // tile_dim1 (tile_dim2 = 0)
  g1[5] = (int)(unsigned)(row_stride_elems & 0xffffffffull); // dim0_stride[31:0]
  g1[6] = (int)(unsigned)((row_stride_elems >> 32) & 0xffffull); // dim0_stride[47:32]
  g1[7] = 0;
  v4i z4 = {0, 0, 0, 0};
  v8i z8 = {0, 0, 0, 0, 0, 0, 0, 0};
  // clang-23 / therock-10.0 arity: (g0, g1, g2, g3, g4, cpol)
  __builtin_amdgcn_tensor_load_to_lds(g0, g1, z4, z4, z8, 0);
}
#endif

// ---- Generic WMMA GEMM:  C[M,N] = A[M,K] @ W[N,K]^T + bias, f32 out ----------
__global__ void __launch_bounds__(256) k_gemm_bias_f32(
    const _Float16* __restrict__ A, const _Float16* __restrict__ W,
    const float* __restrict__ bias, float* __restrict__ C,
    int M, int N, int K)
{
  int w    = (int)(threadIdx.x >> 5);
  int tile = (int)blockIdx.x * 8 + w;
  int tn   = N >> 4;
  int mt   = tile / tn;
  int nt   = tile - mt * tn;
  if ((mt << 4) >= M) return;                    // wave-uniform
  const _Float16* Ab = A + (size_t)(mt << 4) * K;
  const _Float16* Wb = W + (size_t)(nt << 4) * K;
  v8f acc = {0.f, 0.f, 0.f, 0.f, 0.f, 0.f, 0.f, 0.f};
  for (int k = 0; k < K; k += 32)
    acc = wmma16(load_A16x32(Ab + k, K), load_B32x16(Wb + k, K), acc);
  int l = lane31(), cc = l & 15, hs = (l >> 4) & 1;
  float bn = bias[(nt << 4) + cc];
  float* Cb = C + (size_t)((mt << 4) + hs * 8) * N + (nt << 4) + cc;
#pragma unroll
  for (int r = 0; r < 8; ++r) Cb[(size_t)r * N] = acc[r] + bn;
}

// Same GEMM but fp16 output (for hoisted proj_V over all timesteps).
__global__ void __launch_bounds__(256) k_gemm_bias_f16(
    const _Float16* __restrict__ A, const _Float16* __restrict__ W,
    const float* __restrict__ bias, _Float16* __restrict__ C,
    int M, int N, int K)
{
  int w    = (int)(threadIdx.x >> 5);
  int tile = (int)blockIdx.x * 8 + w;
  int tn   = N >> 4;
  int mt   = tile / tn;
  int nt   = tile - mt * tn;
  if ((mt << 4) >= M) return;
  const _Float16* Ab = A + (size_t)(mt << 4) * K;
  const _Float16* Wb = W + (size_t)(nt << 4) * K;
  v8f acc = {0.f, 0.f, 0.f, 0.f, 0.f, 0.f, 0.f, 0.f};
  for (int k = 0; k < K; k += 32)
    acc = wmma16(load_A16x32(Ab + k, K), load_B32x16(Wb + k, K), acc);
  int l = lane31(), cc = l & 15, hs = (l >> 4) & 1;
  float bn = bias[(nt << 4) + cc];
  _Float16* Cb = C + (size_t)((mt << 4) + hs * 8) * N + (nt << 4) + cc;
#pragma unroll
  for (int r = 0; r < 8; ++r) Cb[(size_t)r * N] = (_Float16)(acc[r] + bn);
}

// ---- Per-step attention: one workgroup per batch row --------------------------
__global__ void __launch_bounds__(256) k_att(
    const float* __restrict__ PR, const float* __restrict__ PS,
    const _Float16* __restrict__ PV16, const float* __restrict__ wvec,
    const float* __restrict__ bw, const float* __restrict__ h_s,
    _Float16* __restrict__ hts16, int t)
{
  __shared__ float sbase[HH];
  __shared__ float sbeta[LL];
  __shared__ float salpha[LL];
  int b = (int)blockIdx.x;

  const _Float16* pv = PV16 + ((size_t)b * TT + t) * HH;
  for (int h = (int)threadIdx.x; h < HH; h += 256)
    sbase[h] = PR[b * HH + h] + (float)pv[h];
  __syncthreads();

  int w = (int)(threadIdx.x >> 5), l0 = lane31();
  for (int l = w; l < LL; l += 8) {
    const float* ps = PS + ((size_t)b * LL + l) * HH;
    float s = 0.f;
    for (int h = l0; h < HH; h += 32)
      s += wvec[h] * tanhf(sbase[h] + ps[h]);
#pragma unroll
    for (int off = 16; off; off >>= 1) s += __shfl_xor(s, off, 32);
    if (l0 == 0) sbeta[l] = s + bw[0];
  }
  __syncthreads();

  if (threadIdx.x == 0) {
    float mx = sbeta[0];
    for (int i = 1; i < LL; ++i) mx = fmaxf(mx, sbeta[i]);
    float sum = 0.f;
    for (int i = 0; i < LL; ++i) { float e = expf(sbeta[i] - mx); salpha[i] = e; sum += e; }
    float inv = 1.f / sum;
    for (int i = 0; i < LL; ++i) salpha[i] *= inv;
  }
  __syncthreads();

  for (int d = (int)threadIdx.x; d < HH; d += 256) {
    const float* hs = h_s + (size_t)b * LL * HH + d;
    float acc = 0.f;
#pragma unroll 4
    for (int l = 0; l < LL; ++l) acc += salpha[l] * hs[(size_t)l * HH];
    hts16[b * HH + d] = (_Float16)acc;
  }
}

// ---- Fused gate GEMM + LSTM cell update --------------------------------------
// grid: 4 row-groups (16 batch rows) x 8 col-groups (64 H cols). 512 threads.
// sA = three contiguous 16x512 f16 blocks [x_t | h_att | h], staged by the
// Tensor Data Mover (3 descriptors, one wave issues, s_wait_tensorcnt). 16 waves
// (gate x ntile) run 48 WMMAs each over the 3 K segments; i/f/g/o tiles are
// exchanged via LDS for the fused cell update.
__global__ void __launch_bounds__(512) k_gate(
    const _Float16* __restrict__ hv16, const _Float16* __restrict__ hts16,
    const _Float16* __restrict__ h16c, const _Float16* __restrict__ Wg,
    const float* __restrict__ bg, const float* __restrict__ c_cur,
    float* __restrict__ c_nxt, _Float16* __restrict__ h16n,
    float* __restrict__ out, int t)
{
  __shared__ _Float16 sA[3 * 16 * 512];       // 48 KB, 3 blocks of 16x512
  float* sD = (float*)sA;                     // reused post-barrier (16 KB)

  int b0 = ((int)blockIdx.x >> 3) * 16;
  int j0 = ((int)blockIdx.x & 7) * 64;

#ifdef USE_TDM
  if (threadIdx.x < 32) {                     // wave 0 drives the TDM
    unsigned base = (unsigned)(uintptr_t)sA;  // LDS byte offset (addr[31:0])
    tdm_load_2d_f16(base,
                    hv16 + ((size_t)b0 * TT + t) * HH,
                    16u, (unsigned)HH, (unsigned long long)TT * HH);
    tdm_load_2d_f16(base + 16u * HH * 2u,
                    hts16 + (size_t)b0 * HH, 16u, (unsigned)HH, HH);
    tdm_load_2d_f16(base + 32u * HH * 2u,
                    h16c + (size_t)b0 * HH, 16u, (unsigned)HH, HH);
    __builtin_amdgcn_s_wait_tensorcnt(0);
  }
  __syncthreads();
#else
  for (int idx = (int)threadIdx.x; idx < 4096; idx += 512) {   // dwords per block
    int r = idx >> 8, col = (idx & 255) << 1;
    *(unsigned*)(sA + r * HH + col) =
        *(const unsigned*)(hv16 + ((size_t)(b0 + r) * TT + t) * HH + col);
    *(unsigned*)(sA + 16 * HH + r * HH + col) =
        *(const unsigned*)(hts16 + (size_t)(b0 + r) * HH + col);
    *(unsigned*)(sA + 32 * HH + r * HH + col) =
        *(const unsigned*)(h16c + (size_t)(b0 + r) * HH + col);
  }
  __syncthreads();
#endif

  int w  = (int)(threadIdx.x >> 5);
  int g  = w >> 2;                  // gate: 0=i 1=f 2=g 3=o
  int nt = w & 3;
  int n0 = g * 512 + j0 + nt * 16;  // row in Wg [2048 x 1536]
  const _Float16* Wb = Wg + (size_t)n0 * 1536;

  v8f acc = {0.f, 0.f, 0.f, 0.f, 0.f, 0.f, 0.f, 0.f};
#pragma unroll
  for (int seg = 0; seg < 3; ++seg) {
    const _Float16* As = sA + seg * 16 * HH;
    const _Float16* Ws = Wb + seg * 512;
    for (int k = 0; k < 512; k += 32) {
      __builtin_prefetch(Ws + k + 512, 0, 0);   // global_prefetch_b8, 1KB ahead
      acc = wmma16(load_A16x32(As + k, HH), load_B32x16(Ws + k, 1536), acc);
    }
  }
  __syncthreads();                  // everyone done reading sA

  {
    int l = lane31(), cc = l & 15, hs = (l >> 4) & 1;
#pragma unroll
    for (int r = 0; r < 8; ++r)
      sD[w * 256 + (r + hs * 8) * 16 + cc] = acc[r];
  }
  __syncthreads();

  // Epilogue: 16 rows x 64 cols, two per thread.
  for (int e = (int)threadIdx.x; e < 1024; e += 512) {
    int r = e >> 6, c = e & 63;
    int nt2 = c >> 4, cc = c & 15;
    float gi = sD[(0 * 4 + nt2) * 256 + r * 16 + cc] + bg[0    + j0 + c];
    float gf = sD[(1 * 4 + nt2) * 256 + r * 16 + cc] + bg[512  + j0 + c];
    float gg = sD[(2 * 4 + nt2) * 256 + r * 16 + cc] + bg[1024 + j0 + c];
    float go = sD[(3 * 4 + nt2) * 256 + r * 16 + cc] + bg[1536 + j0 + c];
    int b = b0 + r, j = j0 + c;
    float cp = c_cur[b * HH + j];
    float cn = sigmf(gf) * cp + sigmf(gi) * tanhf(gg);
    float hn = sigmf(go) * tanhf(cn);
    c_nxt[b * HH + j]  = cn;
    h16n[b * HH + j]   = (_Float16)hn;
    out[((size_t)b * TT + t) * HH + j] = hn;
  }
}

// ---- Prep kernels ------------------------------------------------------------
__global__ void k_f32_to_f16(const float* __restrict__ s, _Float16* __restrict__ d, int n) {
  int i = (int)(blockIdx.x * blockDim.x + threadIdx.x);
  if (i < n) d[i] = (_Float16)s[i];
}
__global__ void k_build_Wg(const float* __restrict__ W_ih, const float* __restrict__ W_hh,
                           _Float16* __restrict__ Wg) {
  int i = (int)(blockIdx.x * blockDim.x + threadIdx.x);
  if (i >= 2048 * 1536) return;
  int r = i / 1536, k = i - r * 1536;
  float v = (k < 1024) ? W_ih[(size_t)r * 1024 + k] : W_hh[(size_t)r * 512 + (k - 1024)];
  Wg[i] = (_Float16)v;
}
__global__ void k_add_bias(const float* a, const float* b, float* o, int n) {
  int i = (int)(blockIdx.x * blockDim.x + threadIdx.x);
  if (i < n) o[i] = a[i] + b[i];
}
__global__ void k_zero_h16(_Float16* p, int n) {
  int i = (int)(blockIdx.x * blockDim.x + threadIdx.x);
  if (i < n) p[i] = (_Float16)0.f;
}
__global__ void k_zero_f32(float* p, int n) {
  int i = (int)(blockIdx.x * blockDim.x + threadIdx.x);
  if (i < n) p[i] = 0.f;
}

// ---- Host orchestration ------------------------------------------------------
extern "C" void kernel_launch(void* const* d_in, const int* in_sizes, int n_in,
                              void* d_out, int out_size, void* d_ws, size_t ws_size,
                              hipStream_t stream) {
  (void)in_sizes; (void)n_in; (void)out_size; (void)ws_size;
  const float* h_v  = (const float*)d_in[0];
  const float* h_s  = (const float*)d_in[1];
  const float* W_S  = (const float*)d_in[2];
  const float* b_S  = (const float*)d_in[3];
  const float* W_V  = (const float*)d_in[4];
  const float* b_V  = (const float*)d_in[5];
  const float* W_R  = (const float*)d_in[6];
  const float* b_R  = (const float*)d_in[7];
  const float* W_w  = (const float*)d_in[8];
  const float* b_w  = (const float*)d_in[9];
  const float* W_ih = (const float*)d_in[10];
  const float* b_ih = (const float*)d_in[11];
  const float* W_hh = (const float*)d_in[12];
  const float* b_hh = (const float*)d_in[13];
  float* out = (float*)d_out;

  // Workspace carve (256B aligned)
  char* p = (char*)d_ws;
  auto carve = [&](size_t bytes) { char* r = p; p += (bytes + 255) & ~(size_t)255; return r; };
  _Float16* hv16  = (_Float16*)carve((size_t)BB * TT * HH * 2);   // 33.5 MB
  _Float16* PV16  = (_Float16*)carve((size_t)BB * TT * HH * 2);   // 33.5 MB
  _Float16* hs16  = (_Float16*)carve((size_t)BB * LL * HH * 2);   // 3.1 MB
  _Float16* Wg16  = (_Float16*)carve((size_t)2048 * 1536 * 2);    // 6.3 MB
  _Float16* WR16  = (_Float16*)carve((size_t)HH * HH * 2);
  _Float16* WV16  = (_Float16*)carve((size_t)HH * HH * 2);
  _Float16* WS16  = (_Float16*)carve((size_t)HH * HH * 2);
  float*    PS    = (float*)carve((size_t)BB * LL * HH * 4);      // 6.3 MB
  float*    bg    = (float*)carve((size_t)2048 * 4);
  _Float16* hts16 = (_Float16*)carve((size_t)BB * HH * 2);
  _Float16* h16   = (_Float16*)carve((size_t)2 * BB * HH * 2);    // double buffer
  float*    cbuf  = (float*)carve((size_t)2 * BB * HH * 4);       // double buffer
  float*    PR    = (float*)carve((size_t)BB * HH * 4);

  // Precision prep
  { int n = BB * TT * HH; k_f32_to_f16<<<(n + 255) / 256, 256, 0, stream>>>(h_v, hv16, n); }
  { int n = BB * LL * HH; k_f32_to_f16<<<(n + 255) / 256, 256, 0, stream>>>(h_s, hs16, n); }
  { int n = HH * HH;
    k_f32_to_f16<<<(n + 255) / 256, 256, 0, stream>>>(W_R, WR16, n);
    k_f32_to_f16<<<(n + 255) / 256, 256, 0, stream>>>(W_V, WV16, n);
    k_f32_to_f16<<<(n + 255) / 256, 256, 0, stream>>>(W_S, WS16, n); }
  { int n = 2048 * 1536; k_build_Wg<<<(n + 255) / 256, 256, 0, stream>>>(W_ih, W_hh, Wg16); }
  k_add_bias<<<8, 256, 0, stream>>>(b_ih, b_hh, bg, 2048);
  { int n = 2 * BB * HH;
    k_zero_h16<<<(n + 255) / 256, 256, 0, stream>>>(h16, n);
    k_zero_f32<<<(n + 255) / 256, 256, 0, stream>>>(cbuf, n); }

  // Loop-invariant GEMMs (WMMA): proj_S [B*L,H] and proj_V for ALL timesteps.
  k_gemm_bias_f32<<<(BB * LL / 16) * (HH / 16) / 8, 256, 0, stream>>>(
      hs16, WS16, b_S, PS, BB * LL, HH, HH);
  k_gemm_bias_f16<<<(BB * TT / 16) * (HH / 16) / 8, 256, 0, stream>>>(
      hv16, WV16, b_V, PV16, BB * TT, HH, HH);

  // Sequential scan over T
  for (int t = 0; t < TT; ++t) {
    int cur = t & 1, nxt = cur ^ 1;
    // proj_R = h @ W_R^T + b_R (WMMA, 128 tiles)
    k_gemm_bias_f32<<<(BB / 16) * (HH / 16) / 8, 256, 0, stream>>>(
        h16 + (size_t)cur * BB * HH, WR16, b_R, PR, BB, HH, HH);
    // attention: beta -> softmax -> attended text
    k_att<<<BB, 256, 0, stream>>>(PR, PS, PV16, W_w, b_w, h_s, hts16, t);
    // fused gate GEMM (K=1536) + LSTM cell update
    k_gate<<<32, 512, 0, stream>>>(
        hv16, hts16, h16 + (size_t)cur * BB * HH, Wg16, bg,
        cbuf + (size_t)cur * BB * HH, cbuf + (size_t)nxt * BB * HH,
        h16 + (size_t)nxt * BB * HH, out, t);
  }
}
